// SAGE_47416438947868
// MI455X (gfx1250) — matrix operand order repaired and checked by reading, not compiled, not run on hardware
//
#include <hip/hip_runtime.h>

// ---------------------------------------------------------------------------
// GraphSAGE 3-layer inference for MI455X (gfx1250, wave32, WMMA).
//   layer: h = act( mean_agg(x)[dst] @ Wl + x @ Wr + b )
// Aggregation: fp32 atomic scatter-add (HBM/atomic bound, ~4GB traffic).
// GEMM: v_wmma_f32_16x16x32_bf16, both terms fused, bias+ReLU epilogue.
// Final: wave-per-node log_softmax over 47 logits.
// ---------------------------------------------------------------------------

typedef __attribute__((ext_vector_type(16))) __bf16 v16bf;
typedef __attribute__((ext_vector_type(8)))  float  v8f;

union BF16x16 { v16bf v; uint4 q[2]; };

#define N_NODES 100000
#define N_EDGES 1600000

static __device__ __forceinline__ unsigned short f2bf(float f) {
  unsigned u = __float_as_uint(f);
  u += 0x7fffu + ((u >> 16) & 1u);           // round-to-nearest-even
  return (unsigned short)(u >> 16);
}
static __device__ __forceinline__ float bf2f(unsigned short h) {
  return __uint_as_float(((unsigned)h) << 16);
}

// ---------------- degree count: deg[dst[e]] += 1 ---------------------------
__global__ void deg_count(const int* __restrict__ dst, float* __restrict__ deg, int E) {
  int e = blockIdx.x * blockDim.x + threadIdx.x;
  if (e < E) atomicAdd(&deg[dst[e]], 1.0f);
}

// ---------------- f32 -> bf16 convert (4 elems/thread) ---------------------
__global__ void cvt_f32_bf16(const float* __restrict__ in, unsigned short* __restrict__ out, int nquads) {
  int i = blockIdx.x * blockDim.x + threadIdx.x;
  if (i >= nquads) return;
  float4 v = ((const float4*)in)[i];
  uint2 o;
  o.x = (unsigned)f2bf(v.x) | ((unsigned)f2bf(v.y) << 16);
  o.y = (unsigned)f2bf(v.z) | ((unsigned)f2bf(v.w) << 16);
  ((uint2*)out)[i] = o;
}

// ---------------- weight transpose+convert: Wt[n*K+k] = bf16(W[k*N+n]) -----
__global__ void transpose_w(const float* __restrict__ W, unsigned short* __restrict__ Wt,
                            int K, int N, int Npad) {
  int i = blockIdx.x * blockDim.x + threadIdx.x;
  if (i >= Npad * K) return;
  int n = i / K, k = i % K;
  float v = (n < N) ? W[(size_t)k * N + n] : 0.0f;
  Wt[i] = f2bf(v);
}

// ---------------- scatter-add, f32 source (layer 1) ------------------------
__global__ void scatter_f32(const float* __restrict__ x, const int* __restrict__ src,
                            const int* __restrict__ dst, float* __restrict__ agg,
                            int d4, int total) {
  int i = blockIdx.x * blockDim.x + threadIdx.x;
  if (i >= total) return;
  int e = i / d4, q = i % d4;
  int s = src[e], t = dst[e];
  float4 v = ((const float4*)x)[(size_t)s * d4 + q];
  float* o = agg + ((size_t)t * d4 + q) * 4;
  atomicAdd(o + 0, v.x); atomicAdd(o + 1, v.y);
  atomicAdd(o + 2, v.z); atomicAdd(o + 3, v.w);
}

// ---------------- scatter-add, bf16 source (layers 2,3) --------------------
__global__ void scatter_bf16(const unsigned short* __restrict__ xb, const int* __restrict__ src,
                             const int* __restrict__ dst, float* __restrict__ agg,
                             int d4, int total) {
  int i = blockIdx.x * blockDim.x + threadIdx.x;
  if (i >= total) return;
  int e = i / d4, q = i % d4;
  int s = src[e], t = dst[e];
  uint2 v = ((const uint2*)xb)[(size_t)s * d4 + q];
  float* o = agg + ((size_t)t * d4 + q) * 4;
  atomicAdd(o + 0, __uint_as_float((v.x & 0xffffu) << 16));
  atomicAdd(o + 1, __uint_as_float(v.x & 0xffff0000u));
  atomicAdd(o + 2, __uint_as_float((v.y & 0xffffu) << 16));
  atomicAdd(o + 3, __uint_as_float(v.y & 0xffff0000u));
}

// ---------------- agg / max(deg,1) -> bf16 ---------------------------------
__global__ void norm_to_bf16(const float* __restrict__ aggF, const float* __restrict__ deg,
                             unsigned short* __restrict__ aggB, int d4, int total) {
  int i = blockIdx.x * blockDim.x + threadIdx.x;
  if (i >= total) return;
  int node = i / d4;
  float inv = 1.0f / fmaxf(deg[node], 1.0f);
  float4 v = ((const float4*)aggF)[i];
  uint2 o;
  o.x = (unsigned)f2bf(v.x * inv) | ((unsigned)f2bf(v.y * inv) << 16);
  o.y = (unsigned)f2bf(v.z * inv) | ((unsigned)f2bf(v.w * inv) << 16);
  ((uint2*)aggB)[i] = o;
}

// ---------------- fused SAGE GEMM: D = A1@B1^T + A2@B2^T + bias, opt ReLU --
// A1,A2: [M x K] bf16 row-major.  B1,B2: [Npad x K] bf16 (pre-transposed W).
// One 16x16 output tile per wave; K-loop of v_wmma_f32_16x16x32_bf16 pairs.
__global__ void __launch_bounds__(128)
sage_gemm(const __bf16* __restrict__ A1, const __bf16* __restrict__ B1,
          const __bf16* __restrict__ A2, const __bf16* __restrict__ B2,
          const float* __restrict__ bias,
          float* __restrict__ outF, unsigned short* __restrict__ outB,
          int M, int K, int Nout, int relu, int ldF) {
  const int wave = threadIdx.x >> 5;
  const int lane = threadIdx.x & 31;
  const int half = lane >> 4;
  const int l16  = lane & 15;
  const int mTile = blockIdx.x * 4 + wave;
  if (mTile * 16 >= M) return;                       // wave-uniform: EXEC stays full
  const int nBase = blockIdx.y * 16;
  const int m = mTile * 16 + l16;                    // A row (lanes 0-15 and 16-31 both M=l16)
  const int n = nBase + l16;                         // B column

  const __bf16* a1p = A1 + (size_t)m * K;
  const __bf16* a2p = A2 + (size_t)m * K;
  const __bf16* b1p = B1 + (size_t)n * K;
  const __bf16* b2p = B2 + (size_t)n * K;

  v8f acc = {};
  for (int k = 0; k < K; k += 32) {
    // A layout: elems 0-15 -> K=k+half*16..+15 (contiguous, matches B)
    const int ka = k + half * 16;
    // B layout (same k order as A): elems 0-15 -> K=k+half*16..+15 (contiguous)
    const int kb = k + half * 16;
    BF16x16 a1, a2, b1, b2;
    a1.q[0] = *(const uint4*)(a1p + ka);      a1.q[1] = *(const uint4*)(a1p + ka + 8);
    a2.q[0] = *(const uint4*)(a2p + ka);      a2.q[1] = *(const uint4*)(a2p + ka + 8);
    b1.q[0] = *(const uint4*)(b1p + kb);      b1.q[1] = *(const uint4*)(b1p + kb + 8);
    b2.q[0] = *(const uint4*)(b2p + kb);      b2.q[1] = *(const uint4*)(b2p + kb + 8);
    acc = __builtin_amdgcn_wmma_f32_16x16x32_bf16(false, a1.v, false, b1.v, (short)0, acc, false, false);
    acc = __builtin_amdgcn_wmma_f32_16x16x32_bf16(false, a2.v, false, b2.v, (short)0, acc, false, false);
  }

  const float bv = (n < Nout) ? bias[n] : 0.0f;
  // C/D layout: VGPR v -> row mTile*16 + half*8 + v, col = n
  #pragma unroll
  for (int v = 0; v < 8; ++v) {
    int row = mTile * 16 + half * 8 + v;
    float val = acc[v] + bv;
    if (relu) val = fmaxf(val, 0.0f);
    if (n < Nout) {
      if (outB) outB[(size_t)row * Nout + n] = f2bf(val);
      if (outF) outF[(size_t)row * ldF + n] = val;
    }
  }
}

// ---------------- log_softmax over 47 cols, one wave per node --------------
__global__ void __launch_bounds__(128)
logsoftmax47(const float* __restrict__ H, float* __restrict__ out, int n_nodes) {
  int wave = threadIdx.x >> 5, lane = threadIdx.x & 31;
  int node = blockIdx.x * 4 + wave;
  if (node >= n_nodes) return;
  const float* row = H + (size_t)node * 48;          // ld = 48 (padded)
  float v0 = (lane      < 47) ? row[lane]      : -3.402823466e38f;
  float v1 = (lane + 32 < 47) ? row[lane + 32] : -3.402823466e38f;
  float m = fmaxf(v0, v1);
  for (int off = 16; off; off >>= 1) m = fmaxf(m, __shfl_xor(m, off, 32));
  float s = ((lane < 47) ? expf(v0 - m) : 0.0f) + ((lane + 32 < 47) ? expf(v1 - m) : 0.0f);
  for (int off = 16; off; off >>= 1) s += __shfl_xor(s, off, 32);
  float l = m + logf(s);
  float* orow = out + (size_t)node * 47;
  if (lane      < 47) orow[lane]      = v0 - l;
  if (lane + 32 < 47) orow[lane + 32] = v1 - l;
}

// ---------------------------------------------------------------------------
extern "C" void kernel_launch(void* const* d_in, const int* in_sizes, int n_in,
                              void* d_out, int out_size, void* d_ws, size_t ws_size,
                              hipStream_t stream) {
  const float* x   = (const float*)d_in[0];
  const int*   src = (const int*)  d_in[1];
  const int*   dst = (const int*)  d_in[2];
  const float* Wl0 = (const float*)d_in[3];
  const float* Wr0 = (const float*)d_in[4];
  const float* b0  = (const float*)d_in[5];
  const float* Wl1 = (const float*)d_in[6];
  const float* Wr1 = (const float*)d_in[7];
  const float* b1  = (const float*)d_in[8];
  const float* Wl2 = (const float*)d_in[9];
  const float* Wr2 = (const float*)d_in[10];
  const float* b2  = (const float*)d_in[11];
  float* out = (float*)d_out;

  const int N = N_NODES, E = N_EDGES;

  // ---- carve workspace (all offsets 256B aligned) ----
  char* base = (char*)d_ws;
  size_t off = 0;
  auto carve = [&](size_t bytes) -> char* {
    char* r = base + off;
    off = (off + bytes + 255) & ~(size_t)255;
    return r;
  };
  float*          deg  = (float*)         carve((size_t)N * 4);
  unsigned short* xB   = (unsigned short*)carve((size_t)N * 128 * 2);
  float*          aggF = (float*)         carve((size_t)N * 256 * 4);
  unsigned short* aggB = (unsigned short*)carve((size_t)N * 256 * 2);
  unsigned short* H1B  = (unsigned short*)carve((size_t)N * 256 * 2);
  unsigned short* H2B  = (unsigned short*)carve((size_t)N * 256 * 2);
  float*          H3   = (float*)         carve((size_t)N * 48 * 4);
  unsigned short* Wl0T = (unsigned short*)carve((size_t)256 * 128 * 2);
  unsigned short* Wr0T = (unsigned short*)carve((size_t)256 * 128 * 2);
  unsigned short* Wl1T = (unsigned short*)carve((size_t)256 * 256 * 2);
  unsigned short* Wr1T = (unsigned short*)carve((size_t)256 * 256 * 2);
  unsigned short* Wl2T = (unsigned short*)carve((size_t)48 * 256 * 2);
  unsigned short* Wr2T = (unsigned short*)carve((size_t)48 * 256 * 2);
  (void)ws_size; (void)n_in; (void)in_sizes; (void)out_size;

  const int TB = 256;
  const int MT = (N / 16 + 3) / 4;   // 1563 blocks of 4 waves (6250 M-tiles)

  // ---- degree (graph-invariant across layers) ----
  hipMemsetAsync(deg, 0, (size_t)N * 4, stream);
  deg_count<<<(E + TB - 1) / TB, TB, 0, stream>>>(dst, deg, E);

  // ---- x -> bf16 ----
  cvt_f32_bf16<<<((N * 32) + TB - 1) / TB, TB, 0, stream>>>(x, xB, N * 32);

  // ---- weight transposes (bf16, zero-padded N for layer 3) ----
  transpose_w<<<(256 * 128 + TB - 1) / TB, TB, 0, stream>>>(Wl0, Wl0T, 128, 256, 256);
  transpose_w<<<(256 * 128 + TB - 1) / TB, TB, 0, stream>>>(Wr0, Wr0T, 128, 256, 256);
  transpose_w<<<(256 * 256 + TB - 1) / TB, TB, 0, stream>>>(Wl1, Wl1T, 256, 256, 256);
  transpose_w<<<(256 * 256 + TB - 1) / TB, TB, 0, stream>>>(Wr1, Wr1T, 256, 256, 256);
  transpose_w<<<(48 * 256 + TB - 1) / TB, TB, 0, stream>>>(Wl2, Wl2T, 256, 47, 48);
  transpose_w<<<(48 * 256 + TB - 1) / TB, TB, 0, stream>>>(Wr2, Wr2T, 256, 47, 48);

  // ================= Layer 1 (d_in=128 -> 256, ReLU) =================
  {
    const int d4 = 32, tot = E * d4;
    hipMemsetAsync(aggF, 0, (size_t)N * 128 * 4, stream);
    scatter_f32<<<(tot + TB - 1) / TB, TB, 0, stream>>>(x, src, dst, aggF, d4, tot);
    norm_to_bf16<<<(N * d4 + TB - 1) / TB, TB, 0, stream>>>(aggF, deg, aggB, d4, N * d4);
    dim3 g(MT, 16);
    sage_gemm<<<g, 128, 0, stream>>>((const __bf16*)aggB, (const __bf16*)Wl0T,
                                     (const __bf16*)xB,   (const __bf16*)Wr0T,
                                     b0, nullptr, H1B, N, 128, 256, 1, 0);
  }

  // ================= Layer 2 (256 -> 256, ReLU) =================
  {
    const int d4 = 64, tot = E * d4;
    hipMemsetAsync(aggF, 0, (size_t)N * 256 * 4, stream);
    scatter_bf16<<<(tot + TB - 1) / TB, TB, 0, stream>>>(H1B, src, dst, aggF, d4, tot);
    norm_to_bf16<<<(N * d4 + TB - 1) / TB, TB, 0, stream>>>(aggF, deg, aggB, d4, N * d4);
    dim3 g(MT, 16);
    sage_gemm<<<g, 128, 0, stream>>>((const __bf16*)aggB, (const __bf16*)Wl1T,
                                     (const __bf16*)H1B,  (const __bf16*)Wr1T,
                                     b1, nullptr, H2B, N, 256, 256, 1, 0);
  }

  // ================= Layer 3 (256 -> 47, no act) + log_softmax =======
  {
    const int d4 = 64, tot = E * d4;
    hipMemsetAsync(aggF, 0, (size_t)N * 256 * 4, stream);
    scatter_bf16<<<(tot + TB - 1) / TB, TB, 0, stream>>>(H2B, src, dst, aggF, d4, tot);
    norm_to_bf16<<<(N * d4 + TB - 1) / TB, TB, 0, stream>>>(aggF, deg, aggB, d4, N * d4);
    dim3 g(MT, 3);                       // 47 cols -> 3 N-tiles (padded weights)
    sage_gemm<<<g, 128, 0, stream>>>((const __bf16*)aggB, (const __bf16*)Wl2T,
                                     (const __bf16*)H2B,  (const __bf16*)Wr2T,
                                     b2, H3, nullptr, N, 256, 47, 0, 48);
    logsoftmax47<<<(N + 3) / 4, 128, 0, stream>>>(H3, out, N);
  }
}
